// SCRNNet_69939247448809
// MI455X (gfx1250) — matrix operand order repaired and checked by reading, not compiled, not run on hardware
//
#include <hip/hip_runtime.h>
#include <hip/hip_bf16.h>
#include <string.h>

// ---------------------------------------------------------------------------
// SCRN on MI455X (gfx1250): WMMA f16->f32 GEMMs + weight-stationary persistent
// recurrence (Vh tile held in 128 VGPRs per wave for all 512 steps).
// ---------------------------------------------------------------------------

typedef __attribute__((ext_vector_type(16))) _Float16 v16h;
typedef __attribute__((ext_vector_type(8)))  float    v8f;

#define ALPHA_F      0.95f
#define ONE_M_ALPHA  0.05f
#define BB   64
#define TT   512
#define DIN  256
#define HID  512
#define CTX  256
#define OUTD 128
#define RECUR_BLOCKS 16   // 16 blocks x 8 waves = 128 waves = 128 tiles

union AFrag { v16h v; unsigned u[8]; _Float16 h[16]; };

// ---- fragment loaders (ISA 7.12.2 layouts, wave32) -------------------------
// A (16x32 f16): lane&15 = row M; lanes 0-15 hold K {0..7,16..23}, lanes 16-31
// hold K {8..15,24..31}; VGPR j = K pair starting at 2j (+8 for j>=4).
__device__ __forceinline__ v16h load_frag_a_f16(const _Float16* A, int lda,
                                                int row0, int k0, int lane) {
  AFrag f;
  const _Float16* rp = A + (size_t)(row0 + (lane & 15)) * lda;
  int kb = k0 + ((lane >> 4) << 3);
#pragma unroll
  for (int j = 0; j < 8; ++j) {
    int kk = kb + 2 * j + ((j >= 4) ? 8 : 0);
    f.u[j] = *(const unsigned*)(rp + kk);
  }
  return f.v;
}

__device__ __forceinline__ v16h load_frag_a_f32(const float* A, int lda,
                                                int row0, int k0, int lane) {
  AFrag f;
  const float* rp = A + (size_t)(row0 + (lane & 15)) * lda;
  int kb = k0 + ((lane >> 4) << 3);
#pragma unroll
  for (int j = 0; j < 8; ++j) {
    int kk = kb + 2 * j + ((j >= 4) ? 8 : 0);
    f.h[2 * j]     = (_Float16)rp[kk];
    f.h[2 * j + 1] = (_Float16)rp[kk + 1];
  }
  return f.v;
}

// B (32x16 f16) sourced from row-major weight W[N][K]: lane&15 = col N;
// lanes 0-15 hold K {0..7,16..23}, lanes 16-31 hold K {8..15,24..31}.
__device__ __forceinline__ v16h load_frag_b(const _Float16* W, int ldw,
                                            int col0, int k0, int lane) {
  AFrag f;
  const _Float16* rp = W + (size_t)(col0 + (lane & 15)) * ldw;
  int kb = k0 + ((lane >> 4) << 3);
#pragma unroll
  for (int j = 0; j < 8; ++j)
    f.u[j] = *(const unsigned*)(rp + kb + 2 * j + ((j >= 4) ? 8 : 0));
  return f.v;
}

#define WMMA_F16(a, b, c) \
  __builtin_amdgcn_wmma_f32_16x16x32_f16(false, (a), false, (b), (short)0, (c), false, false)

// ---- fast tanh on the hardware v_exp_f32 pipe ------------------------------
__device__ __forceinline__ float fast_tanh(float x) {
  float ax = fabsf(x);
  float e  = __builtin_amdgcn_exp2f(ax * -2.8853900817779268f); // exp(-2|x|)
  float r  = (1.0f - e) / (1.0f + e);
  return copysignf(r, x);
}

// ---- f32 -> f16 weight conversion -----------------------------------------
__global__ void cvt_f32_to_f16(const float* __restrict__ src,
                               _Float16* __restrict__ dst, int n) {
  int i = blockIdx.x * blockDim.x + threadIdx.x;
  if (i < n) dst[i] = (_Float16)src[i];
}

// ---- generic WMMA GEMM: C[M,N] = A[M,K] @ W[N,K]^T + bias ------------------
// One wave per 16x16 output tile. A is f16 (A16) or f32 (A32). Output either
// f16 (C16) or f32 (C32). permT>0: output row (b*permT+t) stored at row
// (t*(M/permT)+b), i.e. time-major layout for the recurrence.
__global__ __launch_bounds__(256) void gemm_wmma(
    const _Float16* __restrict__ A16, const float* __restrict__ A32, int lda,
    const _Float16* __restrict__ W, int ldw,
    const float* __restrict__ bias,
    _Float16* __restrict__ C16, float* __restrict__ C32, int ldc,
    int M, int N, int K, int permT) {
  int lane = threadIdx.x & 31;
  int wave = (int)((blockIdx.x * blockDim.x + threadIdx.x) >> 5);
  int tilesN = N >> 4;
  int tM = (wave / tilesN) << 4;
  int tN = (wave % tilesN) << 4;
  if (tM >= M) return;  // wave-uniform exit

  v8f acc = {};
  for (int k0 = 0; k0 < K; k0 += 32) {
    v16h a = A32 ? load_frag_a_f32(A32, lda, tM, k0, lane)
                 : load_frag_a_f16(A16, lda, tM, k0, lane);
    if (k0 + 64 < K) {  // stream-ahead hint on A rows
      if (A32) __builtin_prefetch(A32 + (size_t)(tM + (lane & 15)) * lda + k0 + 64, 0, 1);
      else     __builtin_prefetch(A16 + (size_t)(tM + (lane & 15)) * lda + k0 + 64, 0, 1);
    }
    v16h b = load_frag_b(W, ldw, tN, k0, lane);
    acc = WMMA_F16(a, b, acc);
  }

  int col   = tN + (lane & 15);
  int rbase = tM + ((lane >> 4) << 3);
  int mOverT = permT ? (M / permT) : 0;
  float bv = bias ? bias[col] : 0.0f;
#pragma unroll
  for (int r = 0; r < 8; ++r) {
    float v = acc[r] + bv;
    int row = rbase + r;
    int rowOut = permT ? ((row % permT) * mOverT + row / permT) : row;
    size_t idx = (size_t)rowOut * ldc + col;
    if (C32) C32[idx] = v;
    else     C16[idx] = (_Float16)v;
  }
}

// ---- context scan: c_t = (1-a)*c_{t-1} + a*xp_t (in place, f16) ------------
__global__ void scan_ctx(_Float16* __restrict__ xpc, int T, int D, int nseq) {
  int i = blockIdx.x * blockDim.x + threadIdx.x;
  if (i >= nseq) return;
  int b = i / D, d = i % D;
  _Float16* p = xpc + (size_t)b * T * D + d;
  float c = 0.0f;
  for (int t = 0; t < T; ++t) {
    float xp = (float)p[(size_t)t * D];
    c = ONE_M_ALPHA * c + ALPHA_F * xp;
    p[(size_t)t * D] = (_Float16)c;
  }
}

// ---- zero h-state + barrier ------------------------------------------------
__global__ void init_state(_Float16* __restrict__ hping,
                           _Float16* __restrict__ hpong, int n,
                           unsigned* __restrict__ bar) {
  int i = blockIdx.x * blockDim.x + threadIdx.x;
  if (i < n) { hping[i] = (_Float16)0.0f; hpong[i] = (_Float16)0.0f; }
  if (i < 2) bar[i] = 0u;
}

// ---- persistent recurrence: h_t = tanh(cU_t + h_{t-1} @ Vh^T) --------------
// 16 blocks x 8 waves = 128 waves; each wave owns one 16x16 output tile and
// keeps its Vh tile (16 cols x 512 K) register-stationary: 16 B-fragments in
// 128 VGPRs for all T steps. Inner loop: 16x (A-frag load from L2-hot h + WMMA).
// Per-step device-wide sync via agent-scope acquire/release atomic barrier.
__global__ __launch_bounds__(256) void scrn_recur(
    const _Float16* __restrict__ cu,   // [T][B][HID]  (time-major)
    const _Float16* __restrict__ vh,   // [HID][HID] f16 row-major (N,K)
    _Float16* __restrict__ hping, _Float16* __restrict__ hpong,
    _Float16* __restrict__ hseq,       // [B][T][HID] or nullptr
    unsigned* __restrict__ bar, int T) {
  const int lane = threadIdx.x & 31;
  const int wv   = threadIdx.x >> 5;                    // 0..7
  const int row0 = (wv & 3) << 4;                       // batch-row tile
  const int ncol = blockIdx.x * 32 + ((wv >> 2) << 4);  // global col of tile

  // weight-stationary Vh tile: 16 K-step fragments in VGPRs
  v16h bfr[16];
#pragma unroll
  for (int ks = 0; ks < 16; ++ks)
    bfr[ks] = load_frag_b(vh, HID, ncol, ks * 32, lane);

  unsigned localgen = 0;
  for (int t = 0; t < T; ++t) {
    const _Float16* hp = (t & 1) ? hpong : hping;
    _Float16*       hn = (t & 1) ? hping : hpong;

    v8f acc = {};
#pragma unroll
    for (int ks = 0; ks < 16; ++ks) {
      v16h a = load_frag_a_f16(hp, HID, row0, ks * 32, lane);
      acc = WMMA_F16(a, bfr[ks], acc);
    }

    const int colg = ncol + (lane & 15);
    const int rb   = row0 + ((lane >> 4) << 3);
    const _Float16* cut = cu + (size_t)t * BB * HID;   // contiguous 64KB slab
#pragma unroll
    for (int r = 0; r < 8; ++r) {
      int b = rb + r;
      float y = acc[r] + (float)cut[(size_t)b * HID + colg];
      _Float16 h = (_Float16)fast_tanh(y);
      hn[(size_t)b * HID + colg] = h;
      if (hseq) hseq[((size_t)b * T + t) * HID + colg] = h;
    }

    // ---- grid barrier (16 blocks) ----
    __threadfence();
    __syncthreads();
    if (threadIdx.x == 0) {
      unsigned arr = __hip_atomic_fetch_add(&bar[0], 1u, __ATOMIC_ACQ_REL,
                                            __HIP_MEMORY_SCOPE_AGENT);
      if (arr == RECUR_BLOCKS - 1) {
        __hip_atomic_store(&bar[0], 0u, __ATOMIC_RELAXED, __HIP_MEMORY_SCOPE_AGENT);
        __hip_atomic_fetch_add(&bar[1], 1u, __ATOMIC_RELEASE, __HIP_MEMORY_SCOPE_AGENT);
      } else {
        while (__hip_atomic_load(&bar[1], __ATOMIC_ACQUIRE,
                                 __HIP_MEMORY_SCOPE_AGENT) <= localgen) {}
      }
    }
    localgen++;
    __syncthreads();
  }
}

// ---------------------------------------------------------------------------
// Host launcher
// ---------------------------------------------------------------------------
static inline int gemm_blocks(int M, int N) {
  long tiles = (long)(M / 16) * (N / 16);
  long threads = tiles * 32;
  return (int)((threads + 255) / 256);
}

extern "C" void kernel_launch(void* const* d_in, const int* in_sizes, int n_in,
                              void* d_out, int out_size, void* d_ws, size_t ws_size,
                              hipStream_t stream) {
  const float* x    = (const float*)d_in[0];
  const float* Wx0  = (const float*)d_in[1];
  const float* bx0  = (const float*)d_in[2];
  const float* Uc0  = (const float*)d_in[3];
  const float* Vh0  = (const float*)d_in[4];
  const float* Wx1  = (const float*)d_in[5];
  const float* bx1  = (const float*)d_in[6];
  const float* Uc1  = (const float*)d_in[7];
  const float* Vh1  = (const float*)d_in[8];
  const float* fcw  = (const float*)d_in[9];
  const float* fcb  = (const float*)d_in[10];
  float* out = (float*)d_out;

  const int MBT = BB * TT;  // 32768 rows

  // ---- workspace layout ----
  char* ws = (char*)d_ws;
  size_t off = 0;
  auto take = [&](size_t bytes) { char* p = ws + off; off += (bytes + 255) & ~(size_t)255; return p; };
  _Float16* wWx0 = (_Float16*)take((size_t)CTX * DIN * 2);
  _Float16* wUc0 = (_Float16*)take((size_t)HID * CTX * 2);
  _Float16* wVh0 = (_Float16*)take((size_t)HID * HID * 2);
  _Float16* wWx1 = (_Float16*)take((size_t)CTX * HID * 2);
  _Float16* wUc1 = (_Float16*)take((size_t)HID * CTX * 2);
  _Float16* wVh1 = (_Float16*)take((size_t)HID * HID * 2);
  _Float16* wFc  = (_Float16*)take((size_t)OUTD * HID * 2);
  _Float16* xpc  = (_Float16*)take((size_t)MBT * CTX * 2);   // xp / c (in place)
  _Float16* cu   = (_Float16*)take((size_t)MBT * HID * 2);   // cU, time-major
  _Float16* hseq = (_Float16*)take((size_t)MBT * HID * 2);   // layer-0 h sequence
  _Float16* hping = (_Float16*)take((size_t)BB * HID * 2);
  _Float16* hpong = (_Float16*)take((size_t)BB * HID * 2);
  unsigned* bar   = (unsigned*)take(256);
  if (off > ws_size) return;  // workspace too small: deterministic no-op

  // ---- convert weights to f16 ----
  auto cvt = [&](const float* s, _Float16* d, int n) {
    cvt_f32_to_f16<<<(n + 255) / 256, 256, 0, stream>>>(s, d, n);
  };
  cvt(Wx0, wWx0, CTX * DIN);
  cvt(Uc0, wUc0, HID * CTX);
  cvt(Vh0, wVh0, HID * HID);
  cvt(Wx1, wWx1, CTX * HID);
  cvt(Uc1, wUc1, HID * CTX);
  cvt(Vh1, wVh1, HID * HID);
  cvt(fcw, wFc,  OUTD * HID);

  // ================= layer 0 =================
  gemm_wmma<<<gemm_blocks(MBT, CTX), 256, 0, stream>>>(
      nullptr, x, DIN, wWx0, DIN, bx0, xpc, nullptr, CTX, MBT, CTX, DIN, 0);
  scan_ctx<<<(BB * CTX + 255) / 256, 256, 0, stream>>>(xpc, TT, CTX, BB * CTX);
  gemm_wmma<<<gemm_blocks(MBT, HID), 256, 0, stream>>>(      // cU0, time-major
      xpc, nullptr, CTX, wUc0, CTX, nullptr, cu, nullptr, HID, MBT, HID, CTX, TT);
  init_state<<<(BB * HID + 255) / 256, 256, 0, stream>>>(hping, hpong, BB * HID, bar);
  scrn_recur<<<RECUR_BLOCKS, 256, 0, stream>>>(cu, wVh0, hping, hpong, hseq, bar, TT);

  // ================= layer 1 =================
  gemm_wmma<<<gemm_blocks(MBT, CTX), 256, 0, stream>>>(
      hseq, nullptr, HID, wWx1, HID, bx1, xpc, nullptr, CTX, MBT, CTX, HID, 0);
  scan_ctx<<<(BB * CTX + 255) / 256, 256, 0, stream>>>(xpc, TT, CTX, BB * CTX);
  gemm_wmma<<<gemm_blocks(MBT, HID), 256, 0, stream>>>(      // cU1, time-major
      xpc, nullptr, CTX, wUc1, CTX, nullptr, cu, nullptr, HID, MBT, HID, CTX, TT);
  init_state<<<(BB * HID + 255) / 256, 256, 0, stream>>>(hping, hpong, BB * HID, bar);
  scrn_recur<<<RECUR_BLOCKS, 256, 0, stream>>>(cu, wVh1, hping, hpong, nullptr, bar, TT);

  // ================= head =================
  // final h (t = T-1 odd => last write landed in hping)
  gemm_wmma<<<gemm_blocks(BB, OUTD), 256, 0, stream>>>(
      hping, nullptr, HID, wFc, HID, fcb, nullptr, out, OUTD, BB, OUTD, HID, 0);
}
